// SingleHeadAttention_71622874628694
// MI455X (gfx1250) — matrix operand order repaired and checked by reading, not compiled, not run on hardware
//
#include <hip/hip_runtime.h>

// ---------------------------------------------------------------------------
// CDNA5 (gfx1250) WMMA pipeline for LCA-sparsified single-head attention.
// All GEMMs run through v_wmma_f32_16x16x32_f16 (f16 in, f32 accumulate).
// All operand fragments are assembled with 128-bit loads.
// ---------------------------------------------------------------------------

typedef __attribute__((ext_vector_type(16))) _Float16 v16h;
typedef __attribute__((ext_vector_type(8)))  float    v8f;

union Frag {
    v16h     h;
    unsigned u[8];
};

__device__ __forceinline__ v8f wmma_f16(const Frag& a, const Frag& b, v8f c) {
    return __builtin_amdgcn_wmma_f32_16x16x32_f16(
        /*neg_a=*/false, a.h, /*neg_b=*/false, b.h,
        /*c_mod=*/(short)0, c, /*reuse_a=*/false, /*reuse_b=*/false);
}

// row4 points at a 32-f16 (16-dword, 4-uint4) contiguous row holding K=0..31.
// A-operand (16x32): lane<16 holds K {0..7,16..23} -> uint4 {0,2}; lane>=16 -> {1,3}.
__device__ __forceinline__ void load_a_frag(Frag& f, const uint4* row4, int lane) {
    const int s = (lane < 16) ? 0 : 1;
    uint4 x0 = row4[s];
    uint4 x1 = row4[s + 2];
    f.u[0] = x0.x; f.u[1] = x0.y; f.u[2] = x0.z; f.u[3] = x0.w;
    f.u[4] = x1.x; f.u[5] = x1.y; f.u[6] = x1.z; f.u[7] = x1.w;
}
// B-operand (32x16): lane<16 holds K {0..7,16..23} -> uint4 {0,2}; lane>=16 -> {1,3}.
__device__ __forceinline__ void load_b_frag(Frag& f, const uint4* row4, int lane) {
    const int s = (lane < 16) ? 0 : 1;
    uint4 x0 = row4[s];
    uint4 x1 = row4[s + 2];
    f.u[0] = x0.x; f.u[1] = x0.y; f.u[2] = x0.z; f.u[3] = x0.w;
    f.u[4] = x1.x; f.u[5] = x1.y; f.u[6] = x1.z; f.u[7] = x1.w;
}

// ---------------------------------------------------------------------------
// Elementwise helpers
// ---------------------------------------------------------------------------
__global__ void f32_to_f16_k(const float* __restrict__ in, _Float16* __restrict__ out, int n) {
    int i = blockIdx.x * blockDim.x + threadIdx.x;
    if (i < n) out[i] = (_Float16)in[i];
}

// G = 0.5*(Graw + Graw^T), zero diagonal, emit f16 (G is symmetric afterwards).
__global__ void prep_g_k(const float* __restrict__ Graw, _Float16* __restrict__ Gh, int Dd) {
    int j = blockIdx.x * blockDim.x + threadIdx.x;
    int i = blockIdx.y;
    if (j >= Dd) return;
    float g = (i == j) ? 0.0f
                       : 0.5f * (Graw[(size_t)i * Dd + j] + Graw[(size_t)j * Dd + i]);
    Gh[(size_t)i * Dd + j] = (_Float16)g;
}

// First LCA iteration (v=0, a=0): v = eta*u ; a = soft(v, lam)
__global__ void lca_first_k(const float* __restrict__ u, float* __restrict__ vst,
                            _Float16* __restrict__ a, const float* __restrict__ log_lam, int n) {
    int i = blockIdx.x * blockDim.x + threadIdx.x;
    if (i >= n) return;
    float lam = __expf(log_lam[0]);
    float vn  = 0.1f * u[i];
    vst[i] = vn;
    float av = fabsf(vn) - lam;
    a[i] = (_Float16)((av > 0.0f) ? copysignf(av, vn) : 0.0f);
}

// V transpose: in[b][t][d] (f16) -> out[b][d][t] (f16); 32x32 LDS tiles.
__global__ __launch_bounds__(256) void transpose_v_k(
    const _Float16* __restrict__ in, _Float16* __restrict__ out, int T, int Dd)
{
    __shared__ _Float16 tile[32][33];
    const int b  = blockIdx.z;
    const int t0 = blockIdx.x * 32;
    const int d0 = blockIdx.y * 32;
    const int tx = threadIdx.x;        // 0..31
    const int ty = threadIdx.y;        // 0..7
    for (int r = ty; r < 32; r += 8)
        tile[r][tx] = in[(size_t)(b * T + t0 + r) * Dd + d0 + tx];
    __syncthreads();
    for (int r = ty; r < 32; r += 8)
        out[(size_t)(b * Dd + d0 + r) * T + t0 + tx] = tile[tx][r];
}

// ---------------------------------------------------------------------------
// WMMA GEMM: C[M,N] = A[MxK](f16) @ Bm[NxK](f16)^T, fp32 accumulate.
// Block = 256 threads (8 waves). Block tile 128x128, K-step 32.
// Wave (w>>1) picks a 32-row band, (w&1) picks a 64-col band:
// each wave computes 2 m-tiles x 4 n-tiles = 8 WMMAs per K-step.
// MODE 0: Cf32 = acc   MODE 1: Cf16 = acc
// MODE 2: fused LCA epilogue: dv = U - Vst - acc; Vst += eta*dv; Ch = soft(Vst)
// Dims are multiples of tile sizes (8192 x 1024, K=1024).
// ---------------------------------------------------------------------------
template <int MODE>
__global__ __launch_bounds__(256) void gemm16_k(
    const _Float16* __restrict__ A, int lda,
    const _Float16* __restrict__ Bm, int ldb,
    float* __restrict__ Cf, _Float16* __restrict__ Ch, int ldc, int K,
    const float* __restrict__ U, float* __restrict__ Vst,
    const float* __restrict__ log_lam)
{
    __shared__ __align__(16) _Float16 As[128][32];
    __shared__ __align__(16) _Float16 Bs[128][32];

    const int tid  = threadIdx.x;
    const int w    = tid >> 5;
    const int lane = tid & 31;
    const int m0   = blockIdx.y * 128;
    const int n0   = blockIdx.x * 128;

    const int mwave = (w >> 1) * 32;    // 0,32,64,96
    const int nwave = (w & 1) * 64;     // 0,64

    const int srow  = tid >> 1;         // 0..127
    const int shalf = (tid & 1) * 16;

    const _Float16* agp = A + (size_t)(m0 + srow) * lda + shalf;
    const _Float16* bgp = Bm + (size_t)(n0 + srow) * ldb + shalf;

    v8f acc[2][4] = {};

    for (int kk = 0; kk < K; kk += 32) {
        // Stage A/B tiles: each thread moves 32B of each via 2x b128.
        const uint4* as = (const uint4*)(agp + kk);
        const uint4* bs = (const uint4*)(bgp + kk);
        uint4 a0 = as[0], a1 = as[1];
        uint4 b0 = bs[0], b1 = bs[1];
        if (kk + 32 < K) {   // prefetch next K-step staging lines
            __builtin_prefetch((const void*)(agp + kk + 32), 0, 1);
            __builtin_prefetch((const void*)(bgp + kk + 32), 0, 1);
        }
        *((uint4*)&As[srow][shalf])     = a0;
        *((uint4*)&As[srow][shalf] + 1) = a1;
        *((uint4*)&Bs[srow][shalf])     = b0;
        *((uint4*)&Bs[srow][shalf] + 1) = b1;
        __syncthreads();

        Frag af[2];
        #pragma unroll
        for (int mt = 0; mt < 2; ++mt)
            load_a_frag(af[mt], (const uint4*)&As[mwave + mt * 16 + (lane & 15)][0], lane);

        #pragma unroll
        for (int nt = 0; nt < 4; ++nt) {
            Frag bf;
            load_b_frag(bf, (const uint4*)&Bs[nwave + nt * 16 + (lane & 15)][0], lane);
            #pragma unroll
            for (int mt = 0; mt < 2; ++mt)
                acc[mt][nt] = wmma_f16(af[mt], bf, acc[mt][nt]);
        }
        __syncthreads();
    }

    // Epilogue. C/D layout: element j -> row = j + (lane>=16 ? 8 : 0), col = lane&15.
    const int rowOff = (lane >> 4) * 8;
    const int ncol   = lane & 15;
    float lam = 0.0f;
    if (MODE == 2) lam = __expf(log_lam[0]);

    #pragma unroll
    for (int mt = 0; mt < 2; ++mt) {
        #pragma unroll
        for (int nt = 0; nt < 4; ++nt) {
            #pragma unroll
            for (int j = 0; j < 8; ++j) {
                const int gm = m0 + mwave + mt * 16 + j + rowOff;
                const int gn = n0 + nwave + nt * 16 + ncol;
                const size_t idx = (size_t)gm * ldc + gn;
                const float c = acc[mt][nt][j];
                if (MODE == 0) {
                    Cf[idx] = c;
                } else if (MODE == 1) {
                    Ch[idx] = (_Float16)c;
                } else {
                    float vold = Vst[idx];
                    float dv   = U[idx] - vold - c;
                    float vn   = vold + 0.1f * dv;
                    Vst[idx]   = vn;
                    float av   = fabsf(vn) - lam;
                    Ch[idx]    = (_Float16)((av > 0.0f) ? copysignf(av, vn) : 0.0f);
                }
            }
        }
    }
}

// ---------------------------------------------------------------------------
// Flash attention (causal, single head). Block = 128 threads (4 waves).
// Each block: 16 query rows, loops over 32-key tiles up to the causal bound.
// Wave w owns d-range [w*256, w*256+256) for both the score reduction and
// the p@v output accumulation. Vt is [b][d][t] so V fragments load as b128.
// ---------------------------------------------------------------------------
__global__ __launch_bounds__(128) void flash_attn_k(
    const _Float16* __restrict__ Q, const _Float16* __restrict__ Km,
    const _Float16* __restrict__ Vt, _Float16* __restrict__ O,
    int T, int Dd, float scale)
{
    const int b    = blockIdx.y;
    const int q0   = blockIdx.x * 16;
    const int tid  = threadIdx.x;
    const int w    = tid >> 5;
    const int lane = tid & 31;
    const int ml   = lane & 15;
    const int rowOff = (lane >> 4) * 8;
    const int dbase  = w * 256;

    __shared__ float    sc[16][32];
    __shared__ float    mrow[16], lrow[16], rfac[16];
    __shared__ __align__(16) _Float16 pf[16][32];

    // Preload this wave's q fragments (8 chunks of K=32 along d), b128 loads.
    Frag qf[8];
    {
        const _Float16* qrow = Q + (size_t)(b * T + q0 + ml) * Dd + dbase;
        #pragma unroll
        for (int c = 0; c < 8; ++c)
            load_a_frag(qf[c], (const uint4*)(qrow + c * 32), lane);
    }

    v8f acc[16] = {};

    if (tid < 16) { mrow[tid] = -__builtin_inff(); lrow[tid] = 0.0f; }
    __syncthreads();

    const _Float16* vtb = Vt + (size_t)b * Dd * T;
    const int nkt = (q0 + 15) / 32 + 1;

    for (int kt = 0; kt < nkt; ++kt) {
        const int kbase = kt * 32;

        for (int i = tid; i < 512; i += 128) ((float*)sc)[i] = 0.0f;
        __syncthreads();

        // scores partial: q(16 x 256_w) @ k^T -> accumulate into sc via ds_add
        #pragma unroll
        for (int nt2 = 0; nt2 < 2; ++nt2) {
            v8f s = {};
            const _Float16* krow =
                Km + (size_t)(b * T + kbase + nt2 * 16 + ml) * Dd + dbase;
            #pragma unroll
            for (int c = 0; c < 8; ++c) {
                Frag kf;
                load_b_frag(kf, (const uint4*)(krow + c * 32), lane);
                s = wmma_f16(qf[c], kf, s);
            }
            #pragma unroll
            for (int j = 0; j < 8; ++j)
                atomicAdd(&sc[j + rowOff][nt2 * 16 + ml], s[j]);
        }
        __syncthreads();

        // Online softmax: 16 rows handled by threads 0..15.
        if (tid < 16) {
            const int r  = tid;
            const int qg = q0 + r;
            const float mold = mrow[r];
            float sv[32];
            float tmax = -__builtin_inff();
            for (int n = 0; n < 32; ++n) {
                float x = sc[r][n] * scale;
                if (kbase + n > qg) x = -__builtin_inff();
                sv[n] = x;
                tmax = fmaxf(tmax, x);
            }
            const float mnew = fmaxf(mold, tmax);
            const float rf = __expf(mold - mnew);
            float ls = 0.0f;
            for (int n = 0; n < 32; ++n) {
                float e = (sv[n] == -__builtin_inff()) ? 0.0f : __expf(sv[n] - mnew);
                ls += e;
                pf[r][n] = (_Float16)e;
            }
            mrow[r] = mnew;
            lrow[r] = lrow[r] * rf + ls;
            rfac[r] = rf;
        }
        __syncthreads();

        // Rescale accumulators and add p @ v for this wave's d-range.
        float rf2[8];
        #pragma unroll
        for (int j = 0; j < 8; ++j) rf2[j] = rfac[j + rowOff];

        Frag paf;
        load_a_frag(paf, (const uint4*)&pf[ml][0], lane);

        #pragma unroll
        for (int nt = 0; nt < 16; ++nt) {
            #pragma unroll
            for (int j = 0; j < 8; ++j) acc[nt][j] *= rf2[j];

            Frag vf;
            const int d = dbase + nt * 16 + ml;
            load_b_frag(vf, (const uint4*)(vtb + (size_t)d * T + kbase), lane);
            acc[nt] = wmma_f16(paf, vf, acc[nt]);
        }
        __syncthreads();
    }

    // Normalize and store f16 (stores are lane-coalesced f16, 32B per half-wave).
    float rl[8];
    #pragma unroll
    for (int j = 0; j < 8; ++j) rl[j] = lrow[j + rowOff];

    #pragma unroll
    for (int nt = 0; nt < 16; ++nt) {
        const int d = dbase + nt * 16 + ml;
        #pragma unroll
        for (int j = 0; j < 8; ++j) {
            float o = acc[nt][j] / rl[j];
            O[(size_t)(b * T + q0 + j + rowOff) * Dd + d] = (_Float16)o;
        }
    }
}

// ---------------------------------------------------------------------------
// Launch
// ---------------------------------------------------------------------------
extern "C" void kernel_launch(void* const* d_in, const int* in_sizes, int n_in,
                              void* d_out, int out_size, void* d_ws, size_t ws_size,
                              hipStream_t stream) {
    (void)in_sizes; (void)n_in; (void)out_size; (void)ws_size;
    const int Bn = 4, T = 2048, Dd = 1024;
    const int M = Bn * T;                       // 8192
    const int n1 = M * Dd;                      // 8,388,608

    const float* x    = (const float*)d_in[0];
    const float* Wqkv = (const float*)d_in[1];
    const float* Wout = (const float*)d_in[2];
    const float* Gq   = (const float*)d_in[3];
    const float* llq  = (const float*)d_in[4];
    const float* Gk   = (const float*)d_in[5];
    const float* llk  = (const float*)d_in[6];
    float* out = (float*)d_out;

    char* p = (char*)d_ws;
    auto take = [&](size_t bytes) -> char* {
        char* r = p;
        p += (bytes + 255) & ~(size_t)255;
        return r;
    };

    _Float16* x_h    = (_Float16*)take((size_t)n1 * 2);
    _Float16* wqkv_h = (_Float16*)take((size_t)3 * Dd * Dd * 2);
    _Float16* wout_h = (_Float16*)take((size_t)Dd * Dd * 2);
    _Float16* gq_h   = (_Float16*)take((size_t)Dd * Dd * 2);
    _Float16* gk_h   = (_Float16*)take((size_t)Dd * Dd * 2);
    float*    u_q    = (float*)take((size_t)n1 * 4);
    float*    u_k    = (float*)take((size_t)n1 * 4);
    float*    vst_q  = (float*)take((size_t)n1 * 4);
    float*    vst_k  = (float*)take((size_t)n1 * 4);
    _Float16* v_h    = (_Float16*)take((size_t)n1 * 2);
    _Float16* vt_h   = (_Float16*)take((size_t)n1 * 2);
    _Float16* aq[2]  = { (_Float16*)take((size_t)n1 * 2), (_Float16*)take((size_t)n1 * 2) };
    _Float16* ak[2]  = { (_Float16*)take((size_t)n1 * 2), (_Float16*)take((size_t)n1 * 2) };
    _Float16* attn_h = (_Float16*)take((size_t)n1 * 2);

    // 1) fp32 -> f16 operand conversion
    f32_to_f16_k<<<(n1 + 255) / 256, 256, 0, stream>>>(x, x_h, n1);
    f32_to_f16_k<<<(3 * Dd * Dd + 255) / 256, 256, 0, stream>>>(Wqkv, wqkv_h, 3 * Dd * Dd);
    f32_to_f16_k<<<(Dd * Dd + 255) / 256, 256, 0, stream>>>(Wout, wout_h, Dd * Dd);
    prep_g_k<<<dim3(Dd / 256, Dd), 256, 0, stream>>>(Gq, gq_h, Dd);
    prep_g_k<<<dim3(Dd / 256, Dd), 256, 0, stream>>>(Gk, gk_h, Dd);

    // 2) qkv projection: q,k -> fp32 u buffers; v -> f16 (then transpose for attn)
    dim3 gg(Dd / 128, M / 128), gb(256);
    gemm16_k<0><<<gg, gb, 0, stream>>>(x_h, Dd, wqkv_h,                       Dd, u_q, nullptr, Dd, Dd, nullptr, nullptr, nullptr);
    gemm16_k<0><<<gg, gb, 0, stream>>>(x_h, Dd, wqkv_h + (size_t)Dd * Dd,     Dd, u_k, nullptr, Dd, Dd, nullptr, nullptr, nullptr);
    gemm16_k<1><<<gg, gb, 0, stream>>>(x_h, Dd, wqkv_h + (size_t)2 * Dd * Dd, Dd, nullptr, v_h, Dd, Dd, nullptr, nullptr, nullptr);
    transpose_v_k<<<dim3(T / 32, Dd / 32, Bn), dim3(32, 8), 0, stream>>>(v_h, vt_h, T, Dd);

    // 3) LCA: iter 1 elementwise, iters 2..10 as WMMA GEMM + fused epilogue
    lca_first_k<<<(n1 + 255) / 256, 256, 0, stream>>>(u_q, vst_q, aq[0], llq, n1);
    lca_first_k<<<(n1 + 255) / 256, 256, 0, stream>>>(u_k, vst_k, ak[0], llk, n1);
    for (int it = 0; it < 9; ++it) {
        const int bi = it & 1, bo = bi ^ 1;
        gemm16_k<2><<<gg, gb, 0, stream>>>(aq[bi], Dd, gq_h, Dd, nullptr, aq[bo], Dd, Dd, u_q, vst_q, llq);
        gemm16_k<2><<<gg, gb, 0, stream>>>(ak[bi], Dd, gk_h, Dd, nullptr, ak[bo], Dd, Dd, u_k, vst_k, llk);
    }
    // 9 iterations: final a lives in buffer 1.

    // 4) causal flash attention (scale = 1/sqrt(1024))
    flash_attn_k<<<dim3(T / 16, Bn), 128, 0, stream>>>(aq[1], ak[1], vt_h, attn_h, T, Dd, 0.03125f);

    // 5) output projection -> fp32 d_out
    gemm16_k<0><<<gg, gb, 0, stream>>>(attn_h, Dd, wout_h, Dd, out, nullptr, Dd, Dd, nullptr, nullptr, nullptr);
}
